// modelGcn_27702539059447
// MI455X (gfx1250) — compile-verified, matrix operand order checked by reading, not run on hardware
//
#include <hip/hip_runtime.h>

typedef __attribute__((ext_vector_type(16))) _Float16 v16h;
typedef __attribute__((ext_vector_type(8)))  float    v8f;

#define BN_EPS 1e-5f

// ---------------- edge scatter: agg[dst] += feat[src] (templated F -> unrolled wide loads) ----
template <int F>
__global__ __launch_bounds__(256) void scatter_kernel(const float* __restrict__ feat,
                                                      const int* __restrict__ src,
                                                      const int* __restrict__ dst,
                                                      float* __restrict__ agg, int E) {
    int e = blockIdx.x * blockDim.x + threadIdx.x;
    if (e >= E) return;
    int s = src[e], d = dst[e];
    const float* fs = feat + (size_t)s * F;
    float*       fd = agg  + (size_t)d * F;
    float v[F];
#pragma unroll
    for (int f = 0; f < F; ++f) v[f] = fs[f];
#pragma unroll
    for (int f = 0; f < F; ++f) atomicAdd(&fd[f], v[f]);
}

// ------- GIN node update: t = relu(relu((x+agg)W1+b1)W2+b2), + BN stats (templated dims) -----
template <int FIN, int FMID, int FOUT>
__global__ __launch_bounds__(256) void gin_node_kernel(const float* __restrict__ cur,
                                                       const float* __restrict__ agg,
                                                       const float* __restrict__ w1,
                                                       const float* __restrict__ b1,
                                                       const float* __restrict__ w2,
                                                       const float* __restrict__ b2,
                                                       float* __restrict__ out,
                                                       float* __restrict__ stats, int N) {
    __shared__ float sSum[16], sSq[16];
    if (threadIdx.x < 16) { sSum[threadIdx.x] = 0.0f; sSq[threadIdx.x] = 0.0f; }
    __syncthreads();
    int i = blockIdx.x * blockDim.x + threadIdx.x;
    if (i < N) {
        float s[FIN], y1[FMID];
#pragma unroll
        for (int f = 0; f < FIN; ++f)
            s[f] = cur[(size_t)i * FIN + f] + agg[(size_t)i * FIN + f];
#pragma unroll
        for (int m = 0; m < FMID; ++m) {
            float acc = b1[m];
#pragma unroll
            for (int f = 0; f < FIN; ++f) acc += s[f] * w1[f * FMID + m];
            y1[m] = fmaxf(acc, 0.0f);
        }
#pragma unroll
        for (int o = 0; o < FOUT; ++o) {
            float acc = b2[o];
#pragma unroll
            for (int m = 0; m < FMID; ++m) acc += y1[m] * w2[m * FOUT + o];
            float v = fmaxf(acc, 0.0f);
            out[(size_t)i * FOUT + o] = v;
            atomicAdd(&sSum[o], v);
            atomicAdd(&sSq[o], v * v);
        }
    }
    __syncthreads();
    if (threadIdx.x < FOUT) {
        atomicAdd(&stats[threadIdx.x],      sSum[threadIdx.x]);
        atomicAdd(&stats[16 + threadIdx.x], sSq[threadIdx.x]);
    }
}

// ------- BN: finalize per-feature affine coeffs (training-mode batch stats, biased var) -------
__global__ void bn_finalize_kernel(float* __restrict__ stats,
                                   const float* __restrict__ gamma,
                                   const float* __restrict__ beta, int N, int F) {
    int f = threadIdx.x;
    if (f >= F) return;
    float invN = 1.0f / (float)N;
    float mean = stats[f] * invN;
    float var  = stats[16 + f] * invN - mean * mean;
    float inv  = rsqrtf(var + BN_EPS);
    float a    = gamma[f] * inv;
    stats[32 + f] = a;
    stats[48 + f] = beta[f] - mean * a;
}

// ------- BN apply: thread per node row -------
template <int F>
__global__ __launch_bounds__(256) void bn_apply_kernel(float* __restrict__ t,
                                                       const float* __restrict__ stats, int N) {
    int i = blockIdx.x * blockDim.x + threadIdx.x;
    if (i >= N) return;
    float* row = t + (size_t)i * F;
#pragma unroll
    for (int f = 0; f < F; ++f) row[f] = row[f] * stats[32 + f] + stats[48 + f];
}

// ------- global mean pool fused with layer-3 BN apply: pooled += a[f]*h + b[f] -------
__global__ __launch_bounds__(256) void pool_bn_kernel(const float* __restrict__ h,
                                                      const float* __restrict__ stats,
                                                      const int* __restrict__ batch,
                                                      float* __restrict__ pooled,
                                                      float* __restrict__ cnt, int N) {
    int i = blockIdx.x * blockDim.x + threadIdx.x;
    if (i >= N) return;
    int g = batch[i];
    const float* row = h + (size_t)i * 10;
#pragma unroll
    for (int f = 0; f < 10; ++f)
        atomicAdd(&pooled[(size_t)g * 10 + f], row[f] * stats[32 + f] + stats[48 + f]);
    atomicAdd(&cnt[g], 1.0f);
}

__global__ void build_z_kernel(const float* __restrict__ pooled, const float* __restrict__ cnt,
                               const float* __restrict__ gx, float* __restrict__ z, int G) {
    int g = blockIdx.x * blockDim.x + threadIdx.x;
    if (g >= G) return;
    float c = fmaxf(cnt[g], 1.0f);
#pragma unroll
    for (int f = 0; f < 10; ++f) z[(size_t)g * 20 + f]      = pooled[(size_t)g * 10 + f] / c;
#pragma unroll
    for (int j = 0; j < 10; ++j) z[(size_t)g * 20 + 10 + j] = gx[(size_t)g * 10 + j];
}

// ---------------- head layer 1: [G x 20] @ [20 x 128] + bias, relu (WMMA) ----------------
// A layout (16-bit 16x32): lanes 0-15 hold K 0..7 / 16..23, lanes 16-31 hold K 8..15 / 24..31.
// B is the N<->M mirror. C/D: 8 VGPRs, lane&15 = N, M = r (+8 for upper half-wave).
__global__ __launch_bounds__(256) void head1_wmma_kernel(const float* __restrict__ z,
                                                         const float* __restrict__ w,
                                                         const float* __restrict__ bias,
                                                         float* __restrict__ out, int G) {
    const int K = 20, NOUT = 128;
    int lane   = threadIdx.x & 31;
    int wid    = threadIdx.x >> 5;        // 0..7 -> n-tile
    int base_m = blockIdx.x * 16;
    int base_n = wid * 16;
    int half   = lane >> 4;
    int mr     = lane & 15;
    const float* arow = z + (size_t)(base_m + mr) * K;
    v16h a, b;
#pragma unroll
    for (int j = 0; j < 8; ++j) {
        int k0 = half * 8 + j;            // 0..15, always < 20
        int k1 = 16 + half * 8 + j;       // 16..31, pad past K=20
        a[j]     = (_Float16)arow[k0];
        a[8 + j] = (k1 < K) ? (_Float16)arow[k1] : (_Float16)0.0f;
        b[j]     = (_Float16)w[k0 * NOUT + base_n + mr];
        b[8 + j] = (k1 < K) ? (_Float16)w[k1 * NOUT + base_n + mr] : (_Float16)0.0f;
    }
    v8f c = {};
    c = __builtin_amdgcn_wmma_f32_16x16x32_f16(false, a, false, b, (short)0, c, false, false);
    int   col = base_n + mr;
    float bb  = bias[col];
#pragma unroll
    for (int r = 0; r < 8; ++r) {
        int row = base_m + (half ? 8 + r : r);
        out[(size_t)row * NOUT + col] = fmaxf(c[r] + bb, 0.0f);
    }
}

// ---------------- head layer 2: [G x 128] @ [128 x 64] + bias, relu (WMMA, unrolled K) -------
__global__ __launch_bounds__(128) void head2_wmma_kernel(const float* __restrict__ h1,
                                                         const float* __restrict__ w,
                                                         const float* __restrict__ bias,
                                                         float* __restrict__ out, int G) {
    const int K = 128, NOUT = 64;
    int lane   = threadIdx.x & 31;
    int wid    = threadIdx.x >> 5;        // 0..3 -> n-tile
    int base_m = blockIdx.x * 16;
    int base_n = wid * 16;
    int half   = lane >> 4;
    int mr     = lane & 15;
    const float* arow = h1 + (size_t)(base_m + mr) * K;
    v8f c = {};
#pragma unroll
    for (int kk = 0; kk < 4; ++kk) {
        int kb = kk * 32;
        v16h a, b;
#pragma unroll
        for (int j = 0; j < 8; ++j) {
            int k0 = kb + half * 8 + j;
            int k1 = kb + 16 + half * 8 + j;
            a[j]     = (_Float16)arow[k0];
            a[8 + j] = (_Float16)arow[k1];
            b[j]     = (_Float16)w[k0 * NOUT + base_n + mr];
            b[8 + j] = (_Float16)w[k1 * NOUT + base_n + mr];
        }
        c = __builtin_amdgcn_wmma_f32_16x16x32_f16(false, a, false, b, (short)0, c, false, false);
    }
    int   col = base_n + mr;
    float bb  = bias[col];
#pragma unroll
    for (int r = 0; r < 8; ++r) {
        int row = base_m + (half ? 8 + r : r);
        out[(size_t)row * NOUT + col] = fmaxf(c[r] + bb, 0.0f);
    }
}

// ---------------- head layer 3: [G x 64] @ [64 x 1] + bias (float4 loads, 256B-aligned rows) --
__global__ __launch_bounds__(256) void head3_kernel(const float* __restrict__ h2,
                                                    const float* __restrict__ w,
                                                    const float* __restrict__ b,
                                                    float* __restrict__ out, int G) {
    int i = blockIdx.x * blockDim.x + threadIdx.x;
    if (i >= G) return;
    const float4* r4 = (const float4*)(h2 + (size_t)i * 64);
    const float4* w4 = (const float4*)w;
    float acc = b[0];
#pragma unroll
    for (int k = 0; k < 16; ++k) {
        float4 rv = r4[k], wv = w4[k];
        acc += rv.x * wv.x + rv.y * wv.y + rv.z * wv.z + rv.w * wv.w;
    }
    out[i] = acc;
}

extern "C" void kernel_launch(void* const* d_in, const int* in_sizes, int n_in,
                              void* d_out, int out_size, void* d_ws, size_t ws_size,
                              hipStream_t stream) {
    const float* x     = (const float*)d_in[0];
    const int*   eidx  = (const int*)d_in[1];
    const int*   batch = (const int*)d_in[2];
    const float* gx    = (const float*)d_in[3];
    const float* w11 = (const float*)d_in[4];  const float* b11 = (const float*)d_in[5];
    const float* w12 = (const float*)d_in[6];  const float* b12 = (const float*)d_in[7];
    const float* w21 = (const float*)d_in[8];  const float* b21 = (const float*)d_in[9];
    const float* w22 = (const float*)d_in[10]; const float* b22 = (const float*)d_in[11];
    const float* w31 = (const float*)d_in[12]; const float* b31 = (const float*)d_in[13];
    const float* w32 = (const float*)d_in[14]; const float* b32 = (const float*)d_in[15];
    const float* g1  = (const float*)d_in[16]; const float* be1 = (const float*)d_in[17];
    const float* g2  = (const float*)d_in[18]; const float* be2 = (const float*)d_in[19];
    const float* g3  = (const float*)d_in[20]; const float* be3 = (const float*)d_in[21];
    const float* fw1 = (const float*)d_in[22]; const float* fb1 = (const float*)d_in[23];
    const float* fw2 = (const float*)d_in[24]; const float* fb2 = (const float*)d_in[25];
    const float* fw3 = (const float*)d_in[26]; const float* fb3 = (const float*)d_in[27];

    const int N = in_sizes[0] / 5;   // 500000
    const int E = in_sizes[1] / 2;   // 8000000
    const int G = in_sizes[3] / 10;  // 8192

    const int* src = eidx;
    const int* dst = eidx + E;

    // workspace layout (floats); all region sizes are multiples of 4 floats (16B aligned)
    float* ws     = (float*)d_ws;
    float* bufA   = ws;                           // N*10
    float* bufB   = bufA   + (size_t)N * 10;      // N*10
    float* agg    = bufB   + (size_t)N * 10;      // N*10
    float* stats  = agg    + (size_t)N * 10;      // 64: sum[16], sumsq[16], coefA[16], coefB[16]
    float* pooled = stats  + 64;                  // G*10
    float* cnt    = pooled + (size_t)G * 10;      // G
    float* z      = cnt    + G;                   // G*20
    float* h1     = z      + (size_t)G * 20;      // G*128
    float* h2     = h1     + (size_t)G * 128;     // G*64  (rows 256B aligned)
    float* outv   = (float*)d_out;

    const int TB = 256;
    dim3 blkE((E + TB - 1) / TB), blkN((N + TB - 1) / TB), blkG((G + TB - 1) / TB);

    // -------- GIN layer 1 (fin=5, fmid=5, fout=5) --------
    hipMemsetAsync(agg, 0, (size_t)N * 5 * sizeof(float), stream);
    hipMemsetAsync(stats, 0, 64 * sizeof(float), stream);
    scatter_kernel<5><<<blkE, TB, 0, stream>>>(x, src, dst, agg, E);
    gin_node_kernel<5, 5, 5><<<blkN, TB, 0, stream>>>(x, agg, w11, b11, w12, b12, bufA, stats, N);
    bn_finalize_kernel<<<1, 16, 0, stream>>>(stats, g1, be1, N, 5);
    bn_apply_kernel<5><<<blkN, TB, 0, stream>>>(bufA, stats, N);

    // -------- GIN layer 2 (fin=5, fmid=10, fout=10) --------
    hipMemsetAsync(agg, 0, (size_t)N * 5 * sizeof(float), stream);
    hipMemsetAsync(stats, 0, 64 * sizeof(float), stream);
    scatter_kernel<5><<<blkE, TB, 0, stream>>>(bufA, src, dst, agg, E);
    gin_node_kernel<5, 10, 10><<<blkN, TB, 0, stream>>>(bufA, agg, w21, b21, w22, b22, bufB, stats, N);
    bn_finalize_kernel<<<1, 16, 0, stream>>>(stats, g2, be2, N, 10);
    bn_apply_kernel<10><<<blkN, TB, 0, stream>>>(bufB, stats, N);

    // -------- GIN layer 3 (fin=10, fmid=10, fout=10); BN-apply fused into pooling --------
    hipMemsetAsync(agg, 0, (size_t)N * 10 * sizeof(float), stream);
    hipMemsetAsync(stats, 0, 64 * sizeof(float), stream);
    scatter_kernel<10><<<blkE, TB, 0, stream>>>(bufB, src, dst, agg, E);
    gin_node_kernel<10, 10, 10><<<blkN, TB, 0, stream>>>(bufB, agg, w31, b31, w32, b32, bufA, stats, N);
    bn_finalize_kernel<<<1, 16, 0, stream>>>(stats, g3, be3, N, 10);

    // -------- global mean pool (fused layer-3 BN) + concat --------
    hipMemsetAsync(pooled, 0, ((size_t)G * 10 + G) * sizeof(float), stream);  // pooled + cnt
    pool_bn_kernel<<<blkN, TB, 0, stream>>>(bufA, stats, batch, pooled, cnt, N);
    build_z_kernel<<<blkG, TB, 0, stream>>>(pooled, cnt, gx, z, G);

    // -------- head MLP: WMMA GEMMs --------
    head1_wmma_kernel<<<dim3(G / 16), 256, 0, stream>>>(z, fw1, fb1, h1, G);   // 8 waves = 8 n-tiles
    head2_wmma_kernel<<<dim3(G / 16), 128, 0, stream>>>(h1, fw2, fb2, h2, G);  // 4 waves = 4 n-tiles
    head3_kernel<<<blkG, TB, 0, stream>>>(h2, fw3, fb3, outv, G);
}
